// GammaModel_8040178778114
// MI455X (gfx1250) — matrix operand order repaired and checked by reading, not compiled, not run on hardware
//
#include <hip/hip_runtime.h>
#include <hip/hip_bf16.h>
#include <math.h>

// ---------------------------------------------------------------------------
// Model constants (from reference)
// ---------------------------------------------------------------------------
#define BB     32
#define LL     4096
#define MM     (BB * LL)        // 131072 token rows
#define DIN    128              // D_INNER
#define DST    12               // D_STATE
#define DCONV  8
#define DTR    2                // DT_RANK
#define XDBL   (DTR + 2 * DST)  // 26
#define NC     64               // scan chunks
#define CL     (LL / NC)        // 64 steps per chunk

typedef _Float16 f16;
typedef __attribute__((ext_vector_type(16))) _Float16 v16h;
typedef __attribute__((ext_vector_type(8)))  _Float16 v8h;
typedef __attribute__((ext_vector_type(8)))  float    v8f;

// ---------------------------------------------------------------------------
// WMMA helpers (CDNA5 wave32, 16x16x32 f16 -> f32 accumulate)
// ---------------------------------------------------------------------------
__device__ __forceinline__ v8f wmma_f16(v16h a, v16h b, v8f c) {
    // (neg_a, A, neg_b, B, c_mod, C, reuse_a, reuse_b)
    return __builtin_amdgcn_wmma_f32_16x16x32_f16(false, a, false, b,
                                                  (short)0, c, false, false);
}

// A operand: row-major f16 [M, ld]; lane<16 -> K {0..7,16..23}+kb, lane>=16 -> K {8..15,24..31}+kb
__device__ __forceinline__ v16h load_a_f16(const f16* A, int row, int ld, int kb, int lane) {
    int ks = kb + ((lane & 16) ? 8 : 0);
    const f16* p = A + (size_t)row * ld + ks;
    union { v16h v; v8h h[2]; } u;
    u.h[0] = *(const v8h*)(p);
    u.h[1] = *(const v8h*)(p + 16);
    return u.v;
}

// A operand from f32 source (convert in-register)
__device__ __forceinline__ v16h load_a_f32(const float* A, int row, int ld, int lane) {
    int ks = (lane & 16) ? 8 : 0;
    const float* p = A + (size_t)row * ld + ks;
    v16h r;
#pragma unroll
    for (int j = 0; j < 8; ++j) { r[j] = (f16)p[j]; r[j + 8] = (f16)p[j + 16]; }
    return r;
}

// B operand: weight w is [N, K] row-major f32 (torch layout). B[k,n] = w[n,k].
// lane<16 -> K {0..7,16..23}+kb, lane>=16 -> K {8..15,24..31}+kb, n = lane%16 within tile.
__device__ __forceinline__ v16h load_b_w(const float* w, int n, int ldk, int kb, int lane) {
    int ks = kb + ((lane & 16) ? 8 : 0);
    const float* p = w + (size_t)n * ldk + ks;
    v16h r;
#pragma unroll
    for (int j = 0; j < 8; ++j) { r[j] = (f16)p[j]; r[j + 8] = (f16)p[j + 16]; }
    return r;
}

// gfx1250 has a hardware v_tanh_f32 — use it when the builtin is declared.
__device__ __forceinline__ float fast_tanh(float x) {
#if __has_builtin(__builtin_amdgcn_tanhf)
    return __builtin_amdgcn_tanhf(x);
#else
    float e = __expf(-2.0f * fabsf(x));
    float t = (1.0f - e) / (1.0f + e);
    return copysignf(t, x);
#endif
}

__device__ __forceinline__ float softplus_f(float x) {
    return (x > 20.0f) ? x : __logf(1.0f + __expf(x));
}

// ---------------------------------------------------------------------------
// Small kernels
// ---------------------------------------------------------------------------
__global__ void k_init(unsigned* smax) {
    if (threadIdx.x == 0 && blockIdx.x == 0) smax[0] = 0u;
}

__global__ void k_max(const float* __restrict__ x, unsigned* __restrict__ smax, int n) {
    __shared__ unsigned red[256];
    unsigned v = 0;
    for (int i = blockIdx.x * blockDim.x + threadIdx.x; i < n; i += gridDim.x * blockDim.x)
        v = max(v, __float_as_uint(x[(size_t)i * 4 + 2]));   // x[:, :, 2], values >= 0
    red[threadIdx.x] = v;
    __syncthreads();
    for (int s = 128; s > 0; s >>= 1) {
        if (threadIdx.x < s) red[threadIdx.x] = max(red[threadIdx.x], red[threadIdx.x + s]);
        __syncthreads();
    }
    if (threadIdx.x == 0) atomicMax(smax, red[0]);
}

__global__ void k_fc0(const float* __restrict__ x, const float* __restrict__ w,
                      const float* __restrict__ b, const float* __restrict__ smax,
                      float* __restrict__ act) {
    int i = blockIdx.x * blockDim.x + threadIdx.x;        // over M*32
    if (i >= MM * 32) return;
    int row = i >> 5, o = i & 31;
    float inv = 1.0f / smax[0];                            // bit-pattern == float (nonneg)
    const float* xr = x + (size_t)row * 4;
    float s0 = xr[0] * (1.0f / 255.0f);
    float s1 = xr[1] * (1.0f / 255.0f);
    float s2 = xr[2] * inv;
    float s3 = xr[3];
    const float* wr = w + o * 4;
    act[i] = s0 * wr[0] + s1 * wr[1] + s2 * wr[2] + s3 * wr[3] + b[o];
}

// ---------------------------------------------------------------------------
// WMMA GEMM kernels
// ---------------------------------------------------------------------------
// t16 = tanh(act @ lin_w^T + lin_b), f16 out.  grid (M/16, 1), block 64 (2 waves = 2 N-tiles)
__global__ void k_lin_tanh(const float* __restrict__ act, const float* __restrict__ w,
                           const float* __restrict__ bias, f16* __restrict__ t16) {
    int lane = threadIdx.x & 31, wv = threadIdx.x >> 5;
    int mt = blockIdx.x, nt = blockIdx.y * 2 + wv;
    int row0 = mt * 16, nloc = lane & 15, n = nt * 16 + nloc;
    v16h a = load_a_f32(act, row0 + nloc, 32, lane);
    v16h bm = load_b_w(w, n, 32, 0, lane);
    v8f c = {};
    c = wmma_f16(a, bm, c);
    float bo = bias[n];
#pragma unroll
    for (int r = 0; r < 8; ++r) {
        int row = row0 + r + ((lane & 16) ? 8 : 0);
        t16[(size_t)row * 32 + n] = (f16)fast_tanh(c[r] + bo);
    }
}

// xz = t16 @ in_w^T : x -> xpre16 [M,128], z -> z16 [M,128].  grid (M/16, 8), block 64
__global__ void k_inproj(const f16* __restrict__ t16, const float* __restrict__ w,
                         f16* __restrict__ xpre, f16* __restrict__ z16) {
    int lane = threadIdx.x & 31, wv = threadIdx.x >> 5;
    int mt = blockIdx.x, nt = blockIdx.y * 2 + wv;          // 0..15
    int row0 = mt * 16, nloc = lane & 15, n = nt * 16 + nloc;
    v16h a = load_a_f16(t16, row0 + nloc, 32, 0, lane);
    v16h bm = load_b_w(w, n, 32, 0, lane);
    v8f c = {};
    c = wmma_f16(a, bm, c);
#pragma unroll
    for (int r = 0; r < 8; ++r) {
        int row = row0 + r + ((lane & 16) ? 8 : 0);
        f16 v = (f16)c[r];
        if (n < DIN) xpre[(size_t)row * DIN + n]        = v;
        else         z16 [(size_t)row * DIN + (n - DIN)] = v;
    }
}

// act = relu(y16 @ out_w^T).  K=128 -> 4 chained WMMAs.  grid (M/16, 1), block 64
__global__ void k_outproj(const f16* __restrict__ y16, const float* __restrict__ w,
                          float* __restrict__ act) {
    int lane = threadIdx.x & 31, wv = threadIdx.x >> 5;
    int mt = blockIdx.x, nt = blockIdx.y * 2 + wv;          // 0..1
    int row0 = mt * 16, nloc = lane & 15, n = nt * 16 + nloc;
    v8f c = {};
#pragma unroll
    for (int kb = 0; kb < DIN; kb += 32) {
        v16h a  = load_a_f16(y16, row0 + nloc, DIN, kb, lane);
        v16h bm = load_b_w(w, n, DIN, kb, lane);
        c = wmma_f16(a, bm, c);
    }
#pragma unroll
    for (int r = 0; r < 8; ++r) {
        int row = row0 + r + ((lane & 16) ? 8 : 0);
        act[(size_t)row * 32 + n] = fmaxf(c[r], 0.0f);
    }
}

// ---------------------------------------------------------------------------
// Conv + xproj
// ---------------------------------------------------------------------------
// causal depthwise conv (K=8) + bias + SiLU.  one thread per (b,l,d)
__global__ void k_conv(const f16* __restrict__ xpre, const float* __restrict__ cw,
                       const float* __restrict__ cb, f16* __restrict__ xs) {
    int i = blockIdx.x * blockDim.x + threadIdx.x;          // over M*128
    if (i >= MM * DIN) return;
    int d = i & (DIN - 1);
    int bl = i >> 7;
    int l = bl & (LL - 1);
    float acc = cb[d];
    const float* wr = cw + d * DCONV;
#pragma unroll
    for (int k = 0; k < DCONV; ++k) {
        int li = l - (DCONV - 1) + k;
        if (li >= 0)
            acc += wr[k] * (float)xpre[(size_t)(bl - (DCONV - 1 - k)) * DIN + d];
    }
    xs[i] = (f16)(acc / (1.0f + __expf(-acc)));             // silu
}

// x_dbl = xs @ xproj_w^T -> dtbc [M, 26] f32
__global__ void k_xproj(const f16* __restrict__ xs, const float* __restrict__ w,
                        float* __restrict__ dtbc) {
    int i = blockIdx.x * blockDim.x + threadIdx.x;          // over M*26
    if (i >= MM * XDBL) return;
    int row = i / XDBL, j = i - row * XDBL;
    const f16*   xr = xs + (size_t)row * DIN;
    const float* wr = w + j * DIN;
    float acc = 0.0f;
#pragma unroll 8
    for (int k = 0; k < DIN; ++k) acc += wr[k] * (float)xr[k];
    dtbc[i] = acc;
}

// ---------------------------------------------------------------------------
// Chunked parallel selective scan (3 passes)
// Per (b,d,s): h_l = a_l*h_{l-1} + x_l with a_l = exp(delta*A), x_l = delta*u*B_l.
// Pass 1: per chunk compute local state hloc and decay product P (h seeded 0).
// Pass 2: serial prefix over NC chunks; write chunk-entry state in-place to hloc.
// Pass 3: replay chunks seeded with entry state, produce y (+u*D)*silu(z).
// ---------------------------------------------------------------------------
__global__ void k_scan1(const float* __restrict__ dtbc, const f16* __restrict__ xs,
                        const float* __restrict__ dtw, const float* __restrict__ dtb,
                        const float* __restrict__ A_log,
                        float* __restrict__ hloc, float* __restrict__ Pbuf) {
    int b = blockIdx.x / NC, c = blockIdx.x - b * NC;
    int d = threadIdx.x;                                    // 0..127
    __shared__ float sm[CL * XDBL];
    size_t rowbase = (size_t)b * LL + (size_t)c * CL;
    const float* src = dtbc + rowbase * XDBL;
    for (int j = d; j < CL * XDBL; j += DIN) sm[j] = src[j];
    __syncthreads();

    float Arow[DST];
#pragma unroll
    for (int s = 0; s < DST; ++s) Arow[s] = -__expf(A_log[d * DST + s]);
    float w0 = dtw[d * 2 + 0], w1 = dtw[d * 2 + 1], bb = dtb[d];
    float h[DST], P[DST];
#pragma unroll
    for (int s = 0; s < DST; ++s) { h[s] = 0.0f; P[s] = 1.0f; }

    for (int t = 0; t < CL; ++t) {
        const float* r = sm + t * XDBL;
        float u = (float)xs[(rowbase + t) * DIN + d];
        float delta = softplus_f(r[0] * w0 + r[1] * w1 + bb);
        float du = delta * u;
#pragma unroll
        for (int s = 0; s < DST; ++s) {
            float dA = __expf(delta * Arow[s]);
            h[s] = dA * h[s] + du * r[2 + s];
            P[s] *= dA;
        }
    }
    size_t o = ((size_t)blockIdx.x * DIN + d) * DST;
#pragma unroll
    for (int s = 0; s < DST; ++s) { hloc[o + s] = h[s]; Pbuf[o + s] = P[s]; }
}

__global__ void k_scan2(float* __restrict__ hloc, const float* __restrict__ Pbuf) {
    int b = blockIdx.x;                                     // grid B, block 128
    int d = threadIdx.x;
    float h[DST];
#pragma unroll
    for (int s = 0; s < DST; ++s) h[s] = 0.0f;
    for (int c = 0; c < NC; ++c) {
        size_t o = (((size_t)b * NC + c) * DIN + d) * DST;
        float hl[DST], P[DST];
#pragma unroll
        for (int s = 0; s < DST; ++s) { hl[s] = hloc[o + s]; P[s] = Pbuf[o + s]; }
#pragma unroll
        for (int s = 0; s < DST; ++s) {
            hloc[o + s] = h[s];                             // entry state for chunk c
            h[s] = P[s] * h[s] + hl[s];                     // advance past chunk c
        }
    }
}

__global__ void k_scan3(const float* __restrict__ dtbc, const f16* __restrict__ xs,
                        const f16* __restrict__ z16, const float* __restrict__ dtw,
                        const float* __restrict__ dtb, const float* __restrict__ A_log,
                        const float* __restrict__ Dp, const float* __restrict__ hloc,
                        f16* __restrict__ y16) {
    int b = blockIdx.x / NC, c = blockIdx.x - b * NC;
    int d = threadIdx.x;
    __shared__ float sm[CL * XDBL];
    size_t rowbase = (size_t)b * LL + (size_t)c * CL;
    const float* src = dtbc + rowbase * XDBL;
    for (int j = d; j < CL * XDBL; j += DIN) sm[j] = src[j];
    __syncthreads();

    float Arow[DST];
#pragma unroll
    for (int s = 0; s < DST; ++s) Arow[s] = -__expf(A_log[d * DST + s]);
    float w0 = dtw[d * 2 + 0], w1 = dtw[d * 2 + 1], bb = dtb[d], dd = Dp[d];
    float h[DST];
    size_t o = ((size_t)blockIdx.x * DIN + d) * DST;
#pragma unroll
    for (int s = 0; s < DST; ++s) h[s] = hloc[o + s];       // chunk entry state

    for (int t = 0; t < CL; ++t) {
        const float* r = sm + t * XDBL;
        size_t idx = (rowbase + t) * DIN + d;
        float u  = (float)xs[idx];
        float zz = (float)z16[idx];
        float delta = softplus_f(r[0] * w0 + r[1] * w1 + bb);
        float du = delta * u, y = 0.0f;
#pragma unroll
        for (int s = 0; s < DST; ++s) {
            float dA = __expf(delta * Arow[s]);
            h[s] = dA * h[s] + du * r[2 + s];
            y += h[s] * r[2 + DST + s];
        }
        y = (y + u * dd) * (zz / (1.0f + __expf(-zz)));
        y16[idx] = (f16)y;
    }
}

// head: out[b, j] = relu(act[b, L-1, :] @ fc1_w^T + fc1_b)
__global__ void k_head(const float* __restrict__ act, const float* __restrict__ w,
                       const float* __restrict__ b, float* __restrict__ out) {
    int i = threadIdx.x;
    if (i >= BB * 2) return;
    int bt = i >> 1, j = i & 1;
    const float* a = act + ((size_t)bt * LL + (LL - 1)) * 32;
    const float* wr = w + j * 32;
    float acc = b[j];
#pragma unroll
    for (int k = 0; k < 32; ++k) acc += a[k] * wr[k];
    out[i] = fmaxf(acc, 0.0f);
}

// ---------------------------------------------------------------------------
// Launcher
// ---------------------------------------------------------------------------
extern "C" void kernel_launch(void* const* d_in, const int* in_sizes, int n_in,
                              void* d_out, int out_size, void* d_ws, size_t ws_size,
                              hipStream_t stream) {
    const float* x      = (const float*)d_in[0];
    const float* fc0_w  = (const float*)d_in[1];
    const float* fc0_b  = (const float*)d_in[2];
    const float* lin_w  = (const float*)d_in[3];
    const float* lin_b  = (const float*)d_in[4];
    const float* in_w   = (const float*)d_in[5];
    const float* conv_w = (const float*)d_in[6];
    const float* conv_b = (const float*)d_in[7];
    const float* xprj_w = (const float*)d_in[8];
    const float* dt_w   = (const float*)d_in[9];
    const float* dt_b   = (const float*)d_in[10];
    const float* A_log  = (const float*)d_in[11];
    const float* Dp     = (const float*)d_in[12];
    const float* out_w  = (const float*)d_in[13];
    const float* fc1_w  = (const float*)d_in[14];
    const float* fc1_b  = (const float*)d_in[15];
    float* out = (float*)d_out;

    // workspace layout (~165 MB total)
    char* p = (char*)d_ws;
    unsigned* s_max = (unsigned*)p;          p += 256;
    float* act   = (float*)p;                p += (size_t)MM * 32 * 4;
    f16*   t16   = (f16*)p;                  p += (size_t)MM * 32 * 2;
    f16*   xpre  = (f16*)p;                  p += (size_t)MM * DIN * 2;  // reused as y16
    f16*   z16   = (f16*)p;                  p += (size_t)MM * DIN * 2;
    f16*   xs16  = (f16*)p;                  p += (size_t)MM * DIN * 2;
    float* dtbc  = (float*)p;                p += (size_t)MM * XDBL * 4;
    float* hloc  = (float*)p;                p += (size_t)BB * NC * DIN * DST * 4;
    float* Pbuf  = (float*)p;                p += (size_t)BB * NC * DIN * DST * 4;
    (void)ws_size; (void)in_sizes; (void)n_in; (void)out_size;

    k_init<<<1, 1, 0, stream>>>(s_max);
    k_max<<<256, 256, 0, stream>>>(x, s_max, MM);
    k_fc0<<<(MM * 32 + 255) / 256, 256, 0, stream>>>(x, fc0_w, fc0_b, (const float*)s_max, act);

    for (int i = 0; i < 3; ++i) {
        const float* lw  = lin_w  + (size_t)i * 32 * 32;
        const float* lb  = lin_b  + (size_t)i * 32;
        const float* iw  = in_w   + (size_t)i * 256 * 32;
        const float* cw  = conv_w + (size_t)i * DIN * DCONV;
        const float* cb  = conv_b + (size_t)i * DIN;
        const float* xw  = xprj_w + (size_t)i * XDBL * DIN;
        const float* dw  = dt_w   + (size_t)i * DIN * DTR;
        const float* db  = dt_b   + (size_t)i * DIN;
        const float* al  = A_log  + (size_t)i * DIN * DST;
        const float* dp  = Dp     + (size_t)i * DIN;
        const float* ow  = out_w  + (size_t)i * 32 * DIN;

        k_lin_tanh<<<dim3(MM / 16, 1), 64, 0, stream>>>(act, lw, lb, t16);
        k_inproj  <<<dim3(MM / 16, 8), 64, 0, stream>>>(t16, iw, xpre, z16);
        k_conv    <<<(MM * DIN + 255) / 256, 256, 0, stream>>>(xpre, cw, cb, xs16);
        k_xproj   <<<(MM * XDBL + 255) / 256, 256, 0, stream>>>(xs16, xw, dtbc);
        k_scan1   <<<BB * NC, DIN, 0, stream>>>(dtbc, xs16, dw, db, al, hloc, Pbuf);
        k_scan2   <<<BB, DIN, 0, stream>>>(hloc, Pbuf);
        k_scan3   <<<BB * NC, DIN, 0, stream>>>(dtbc, xs16, z16, dw, db, al, dp, hloc,
                                                xpre /*y16*/);
        k_outproj <<<dim3(MM / 16, 1), 64, 0, stream>>>(xpre /*y16*/, ow, act);
    }

    k_head<<<1, 64, 0, stream>>>(act, fc1_w, fc1_b, out);
}
